// DecoderNetwork_68968584839852
// MI455X (gfx1250) — compile-verified, matrix operand order checked by reading, not run on hardware
//
#include <hip/hip_runtime.h>

// ---------------------------------------------------------------------------
// MI455X (gfx1250) implementation.
//   * all GEMMs in bf16 with f32 accumulate via v_wmma_f32_16x16x32_bf16
//   * GEMM weight tiles staged to LDS with global_load_async_to_lds_b128
//     (ASYNCcnt), double-buffered to overlap DMA with WMMA
//   * LSTM: x@Wih^T hoisted into one big WMMA GEMM per layer; serial scan only
//     does h@Whh^T per step (4 independent blocks of 16 batch rows)
//   * attention: batched WMMA GEMMs for keys/scores/ctx, wave32 softmax
// ---------------------------------------------------------------------------

typedef __attribute__((ext_vector_type(16))) __bf16 v16bf;
typedef __attribute__((ext_vector_type(8)))  float  v8f;

static constexpr int kB = 64;
static constexpr int kT = 512;
static constexpr int kH = 256;
static constexpr int kO = 128;
static constexpr int k4H = 1024;
static constexpr int kBT = kB * kT;       // 32768

__device__ __forceinline__ unsigned short f2bf(float f) {
  unsigned u = __float_as_uint(f);
  unsigned r = u + 0x7FFFu + ((u >> 16) & 1u);   // round-to-nearest-even
  return (unsigned short)(r >> 16);
}
__device__ __forceinline__ float bf2f(unsigned short h) {
  return __uint_as_float(((unsigned)h) << 16);
}

// Async DMA: copy 16B from global to LDS per lane (GLOBAL_LOAD_ASYNC_TO_LDS_B128,
// tracked by ASYNCcnt). VDST operand carries the LDS byte address; for a generic
// pointer into LDS the low 32 bits are exactly that (ISA 10.2).
__device__ __forceinline__ void async_copy_b128(const unsigned short* g,
                                                unsigned short* l) {
  unsigned loff = (unsigned)(unsigned long long)l;
  asm volatile("global_load_async_to_lds_b128 %0, %1, off"
               :: "v"(loff), "v"(g) : "memory");
}
__device__ __forceinline__ void wait_async_le2() {
  asm volatile("s_wait_asynccnt 0x2" ::: "memory");
}
__device__ __forceinline__ void wait_async_0() {
  asm volatile("s_wait_asynccnt 0x0" ::: "memory");
}

// Load a 16x32 bf16 WMMA fragment (A from [M,K] row-major, or B from [N,K]
// row-major, ld in elements). Per ISA 7.12.2: lane L holds row (L&15),
// elements i: K = (i%8) + 16*(i/8) + 8*(L>>4) -> two contiguous 16B runs.
__device__ __forceinline__ v16bf load_frag16(const unsigned short* p,
                                             int row0, int k0, int ld) {
  int lane = threadIdx.x & 31;
  const unsigned short* base = p + (row0 + (lane & 15)) * ld + k0 + 8 * (lane >> 4);
  union { uint4 u[2]; v16bf v; } cv;
  cv.u[0] = *(const uint4*)(base);        // K + 0..7
  cv.u[1] = *(const uint4*)(base + 16);   // K + 16..23
  return cv.v;
}

// ---------------------------------------------------------------------------
// Generic tiled GEMM: C[M,N] = act(A[M,K] @ W[N,K]^T + bias1 + bias2)
// A, W bf16; C f32 (optional) and/or bf16 (optional). Batched via blockIdx.z.
// Tile: 128x128 per block (8 waves, each wave = 16 rows x 8 WMMA tiles).
// W tile staged in LDS by async DMA, double buffered (128x32 bf16 = 8 KB/buf).
// All problem dims here are multiples of the tile sizes.
// ---------------------------------------------------------------------------
__global__ __launch_bounds__(256) void gemm_nt_kernel(
    const unsigned short* __restrict__ A, const unsigned short* __restrict__ W,
    float* __restrict__ C, unsigned short* __restrict__ Cb,
    const float* __restrict__ bias1, const float* __restrict__ bias2,
    int K, int lda, int ldb, int ldc,
    long long sA, long long sW, long long sC, long long sCb, int act) {
  __shared__ __align__(16) unsigned short wtile[2][128 * 32];
  long long z = blockIdx.z;
  A += z * sA;
  W += z * sW;
  int tid = threadIdx.x;
  int wave = tid >> 5, lane = tid & 31;
  int m0 = blockIdx.y * 128 + wave * 16;
  int nb = blockIdx.x * 128;

  // Stage one 128(N) x 32(K) chunk of W into LDS: 512 x 16B, 2 per thread.
  auto stage = [&](int k0, int buf) {
#pragma unroll
    for (int s = 0; s < 2; ++s) {
      int c = tid * 2 + s;
      int nl = c >> 2, koff = (c & 3) * 8;
      async_copy_b128(W + (nb + nl) * ldb + k0 + koff,
                      &wtile[buf][nl * 32 + koff]);
    }
  };

  v8f acc[8];
  v8f zero = {0.f, 0.f, 0.f, 0.f, 0.f, 0.f, 0.f, 0.f};
#pragma unroll
  for (int j = 0; j < 8; ++j) acc[j] = zero;

  int nk = K / 32;
  stage(0, 0);
  for (int kc = 0; kc < nk; ++kc) {
    if (kc + 1 < nk) {
      stage((kc + 1) * 32, (kc + 1) & 1);  // overlap next DMA with this chunk
      wait_async_le2();                     // current chunk landed (in-order)
    } else {
      wait_async_0();
    }
    __syncthreads();
    v16bf a = load_frag16(A, m0, kc * 32, lda);
    const unsigned short* wt = wtile[kc & 1];
#pragma unroll
    for (int j = 0; j < 8; ++j) {
      v16bf b = load_frag16(wt, j * 16, 0, 32);
      acc[j] = __builtin_amdgcn_wmma_f32_16x16x32_bf16(
          false, a, false, b, (short)0, acc[j], false, false);
    }
    __syncthreads();  // everyone done reading before buffer is overwritten
  }

  int grp = lane >> 4, cn = lane & 15;
#pragma unroll
  for (int j = 0; j < 8; ++j) {
    int col = nb + j * 16 + cn;
    float badd = 0.f;
    if (bias1) badd += bias1[col];
    if (bias2) badd += bias2[col];
#pragma unroll
    for (int r = 0; r < 8; ++r) {
      int row = m0 + r + 8 * grp;
      float v = acc[j][r] + badd;
      if (act == 1) v = tanhf(v);
      long long off = (long long)row * ldc + col;
      if (C)  (C  + z * sC )[off] = v;
      if (Cb) (Cb + z * sCb)[off] = f2bf(v);
    }
  }
}

// ---------------------------------------------------------------------------
// LSTM scan: one block owns 16 batch rows for all T steps (rows independent).
// Per step: g = h @ Whh^T (WMMA, 16 waves x 4 tiles covering N=1024) staged
// to LDS, then gates combine with precomputed xg (x@Wih^T + biases).
// Whh (512 KB bf16) stays L2-resident across the 512 steps.
// ---------------------------------------------------------------------------
__global__ __launch_bounds__(512) void lstm_scan_kernel(
    const float* __restrict__ xg,            // [B,T,4H], biases included
    const unsigned short* __restrict__ whh,  // [4H,H] bf16 layer slice
    const float* __restrict__ h0,            // [B,H] layer slice
    const float* __restrict__ c0,            // [B,H] layer slice
    unsigned short* __restrict__ hall,       // out bf16, row stride ldh
    int ldh) {
  __shared__ __align__(16) unsigned short h_sh[16 * kH];    // 8 KB
  __shared__ unsigned short g_sh[16 * k4H];                 // 32 KB
  int tid = threadIdx.x;
  int b0 = blockIdx.x * 16;

  float c[8];
#pragma unroll
  for (int i = 0; i < 8; ++i) {
    int idx = i * 512 + tid;
    int r = idx >> 8, u = idx & 255;
    h_sh[idx] = f2bf(h0[(b0 + r) * kH + u]);
    c[i] = c0[(b0 + r) * kH + u];
  }

  int wave = tid >> 5, lane = tid & 31;
  int grp = lane >> 4, cn = lane & 15;
  v8f zero = {0.f, 0.f, 0.f, 0.f, 0.f, 0.f, 0.f, 0.f};

  for (int t = 0; t < kT; ++t) {
    __syncthreads();  // h_sh ready for GEMM
    v8f acc[4];
#pragma unroll
    for (int j = 0; j < 4; ++j) acc[j] = zero;
#pragma unroll
    for (int k0 = 0; k0 < kH; k0 += 32) {
      v16bf a = load_frag16(h_sh, 0, k0, kH);
#pragma unroll
      for (int j = 0; j < 4; ++j) {
        v16bf b = load_frag16(whh, wave * 64 + j * 16, k0, kH);
        acc[j] = __builtin_amdgcn_wmma_f32_16x16x32_bf16(
            false, a, false, b, (short)0, acc[j], false, false);
      }
    }
#pragma unroll
    for (int j = 0; j < 4; ++j) {
      int col = wave * 64 + j * 16 + cn;
#pragma unroll
      for (int r = 0; r < 8; ++r) g_sh[(r + 8 * grp) * k4H + col] = f2bf(acc[j][r]);
    }
    __syncthreads();  // g_sh ready for gate pass

#pragma unroll
    for (int i = 0; i < 8; ++i) {
      int idx = i * 512 + tid;
      int r = idx >> 8, u = idx & 255;
      long long base = ((long long)(b0 + r) * kT + t) * k4H + u;
      float gi = bf2f(g_sh[r * k4H + u      ]) + xg[base      ];
      float gf = bf2f(g_sh[r * k4H + u + 256]) + xg[base + 256];
      float gg = bf2f(g_sh[r * k4H + u + 512]) + xg[base + 512];
      float go = bf2f(g_sh[r * k4H + u + 768]) + xg[base + 768];
      float ii = 1.f / (1.f + __expf(-gi));
      float ff = 1.f / (1.f + __expf(-gf));
      float gt = tanhf(gg);
      float oo = 1.f / (1.f + __expf(-go));
      c[i] = ff * c[i] + ii * gt;
      float hh = oo * tanhf(c[i]);
      unsigned short hb = f2bf(hh);
      h_sh[r * kH + u] = hb;
      hall[((long long)(b0 + r) * kT + t) * ldh + u] = hb;
    }
  }
}

// ---------------------------------------------------------------------------
// Elementwise / reduction kernels
// ---------------------------------------------------------------------------
__global__ void cvt_f32_bf16_kernel(const float* __restrict__ s,
                                    unsigned short* __restrict__ d, int n) {
  int i = blockIdx.x * blockDim.x + threadIdx.x;
  if (i < n) d[i] = f2bf(s[i]);
}

__global__ void gather_emb_kernel(const float* __restrict__ emb,
                                  const int* __restrict__ actions,
                                  unsigned short* __restrict__ out, int n) {
  int i = blockIdx.x * blockDim.x + threadIdx.x;
  if (i >= n) return;
  int h = i & (kH - 1);
  int bt = i >> 8;               // b*T + t
  int t = bt & (kT - 1);
  int b = bt >> 9;
  int id = (t == 0) ? 0 : actions[b * kT + t - 1];
  out[i] = f2bf(emb[id * kH + h]);
}

__global__ void transpose_enc_kernel(const float* __restrict__ enc,   // [B,S,H]
                                     unsigned short* __restrict__ et, // [B,H,S]
                                     int n) {
  int i = blockIdx.x * blockDim.x + threadIdx.x;
  if (i >= n) return;
  int s = i & (kT - 1);
  int hs = i >> 9;
  int h = hs & (kH - 1);
  int b = hs >> 8;
  et[i] = f2bf(enc[((long long)b * kT + s) * kH + h]);
}

__global__ void actions_to_f32_kernel(const int* __restrict__ a,
                                      float* __restrict__ o, int n) {
  int i = blockIdx.x * blockDim.x + threadIdx.x;
  if (i < n) o[i] = (float)a[i];
}

// one wave32 per row of 512 scores
__global__ __launch_bounds__(256) void softmax_kernel(
    const float* __restrict__ s, unsigned short* __restrict__ p) {
  int wave = threadIdx.x >> 5, lane = threadIdx.x & 31;
  long long row = blockIdx.x * 8 + wave;
  const float* sr = s + row * kT;
  float v[16];
  float mx = -3.4e38f;
#pragma unroll
  for (int j = 0; j < 16; ++j) { v[j] = sr[lane + j * 32]; mx = fmaxf(mx, v[j]); }
#pragma unroll
  for (int m = 16; m >= 1; m >>= 1) mx = fmaxf(mx, __shfl_xor(mx, m, 32));
  float sum = 0.f;
#pragma unroll
  for (int j = 0; j < 16; ++j) { v[j] = __expf(v[j] - mx); sum += v[j]; }
#pragma unroll
  for (int m = 16; m >= 1; m >>= 1) sum += __shfl_xor(sum, m, 32);
  float inv = 1.f / sum;
  unsigned short* pr = p + row * kT;
#pragma unroll
  for (int j = 0; j < 16; ++j) pr[lane + j * 32] = f2bf(v[j] * inv);
}

// values[row] = sum_o q[row,o] * logits[row,o]; one wave32 per row (O=128)
__global__ __launch_bounds__(256) void values_kernel(
    const float* __restrict__ q, const float* __restrict__ lg,
    float* __restrict__ out) {
  int wave = threadIdx.x >> 5, lane = threadIdx.x & 31;
  long long row = blockIdx.x * 8 + wave;
  const float* qr = q + row * kO;
  const float* lr = lg + row * kO;
  float s = 0.f;
#pragma unroll
  for (int j = 0; j < 4; ++j) { int o = lane + j * 32; s += qr[o] * lr[o]; }
#pragma unroll
  for (int m = 16; m >= 1; m >>= 1) s += __shfl_xor(s, m, 32);
  if (lane == 0) out[row] = s;
}

// ---------------------------------------------------------------------------
// Host launch
// ---------------------------------------------------------------------------
static inline int cdiv(int a, int b) { return (a + b - 1) / b; }

extern "C" void kernel_launch(void* const* d_in, const int* in_sizes, int n_in,
                              void* d_out, int out_size, void* d_ws, size_t ws_size,
                              hipStream_t stream) {
  const float* enc       = (const float*)d_in[0];   // [B,T,H]
  const float* h0        = (const float*)d_in[1];   // [L,B,H]
  const float* c0        = (const float*)d_in[2];
  const int*   actions   = (const int*)  d_in[3];   // [B,T]
  const float* embedding = (const float*)d_in[4];   // [O,H]
  const float* Wih       = (const float*)d_in[5];   // [L,4H,H]
  const float* Whh       = (const float*)d_in[6];
  const float* bih       = (const float*)d_in[7];   // [L,4H]
  const float* bhh       = (const float*)d_in[8];
  const float* W_attn    = (const float*)d_in[9];   // [H,H]
  const float* b_attn    = (const float*)d_in[10];
  const float* W_concat  = (const float*)d_in[11];  // [H,2H]
  const float* b_concat  = (const float*)d_in[12];
  const float* W_out     = (const float*)d_in[13];  // [O,H]
  const float* W_critic  = (const float*)d_in[14];  // [O,O]
  const float* b_critic  = (const float*)d_in[15];
  (void)in_sizes; (void)n_in; (void)out_size; (void)ws_size;

  float* outp        = (float*)d_out;
  float* action_out  = outp;                       // [B*T]
  float* logits_out  = outp + kBT;                 // [B*T,O]
  float* values_out  = outp + kBT + (long long)kBT * kO;

  // ---- workspace carving ----
  char* wp = (char*)d_ws;
  auto allocF = [&](size_t n) { float* p = (float*)wp; wp += n * 4; return p; };
  auto allocH = [&](size_t n) { unsigned short* p = (unsigned short*)wp; wp += n * 2; return p; };

  float* xg      = allocF((size_t)kBT * k4H);   // x-projection gates (reused per layer)
  float* scores  = allocF((size_t)kBT * kT);
  float* qbuf    = allocF((size_t)kBT * kO);
  unsigned short* wih_bf  = allocH(2 * (size_t)k4H * kH);
  unsigned short* whh_bf  = allocH(2 * (size_t)k4H * kH);
  unsigned short* wattn_bf= allocH((size_t)kH * kH);
  unsigned short* wcat_bf = allocH((size_t)kH * 2 * kH);
  unsigned short* wout_bf = allocH((size_t)kO * kH);
  unsigned short* wcrit_bf= allocH((size_t)kO * kO);
  unsigned short* emb_bf  = allocH((size_t)kBT * kH);
  unsigned short* enc_bf  = allocH((size_t)kBT * kH);
  unsigned short* encT_bf = allocH((size_t)kBT * kH);   // [B,H,S]
  unsigned short* h0all_bf= allocH((size_t)kBT * kH);
  unsigned short* keys_bf = allocH((size_t)kBT * kH);
  unsigned short* cat_bf  = allocH((size_t)kBT * 2 * kH); // [out | ctx]
  unsigned short* probs_bf= allocH((size_t)kBT * kT);
  unsigned short* outc_bf = allocH((size_t)kBT * kH);
  unsigned short* logits_bf = allocH((size_t)kBT * kO);

  dim3 blk256(256);
  auto cvt = [&](const float* s, unsigned short* d, int n) {
    cvt_f32_bf16_kernel<<<cdiv(n, 256), blk256, 0, stream>>>(s, d, n);
  };

  // ---- weight / input conversion ----
  cvt(Wih,      wih_bf,   2 * k4H * kH);
  cvt(Whh,      whh_bf,   2 * k4H * kH);
  cvt(W_attn,   wattn_bf, kH * kH);
  cvt(W_concat, wcat_bf,  kH * 2 * kH);
  cvt(W_out,    wout_bf,  kO * kH);
  cvt(W_critic, wcrit_bf, kO * kO);
  cvt(enc,      enc_bf,   kBT * kH);
  gather_emb_kernel<<<cdiv(kBT * kH, 256), blk256, 0, stream>>>(
      embedding, actions, emb_bf, kBT * kH);
  transpose_enc_kernel<<<cdiv(kBT * kH, 256), blk256, 0, stream>>>(
      enc, encT_bf, kBT * kH);

  // ---- LSTM layer 0 ----
  gemm_nt_kernel<<<dim3(k4H / 128, kBT / 128, 1), blk256, 0, stream>>>(
      emb_bf, wih_bf, xg, nullptr, bih, bhh,
      kH, kH, kH, k4H, 0, 0, 0, 0, 0);
  lstm_scan_kernel<<<dim3(4), dim3(512), 0, stream>>>(
      xg, whh_bf, h0, c0, h0all_bf, kH);

  // ---- LSTM layer 1 (output goes into left half of cat buffer) ----
  gemm_nt_kernel<<<dim3(k4H / 128, kBT / 128, 1), blk256, 0, stream>>>(
      h0all_bf, wih_bf + (size_t)k4H * kH, xg, nullptr, bih + k4H, bhh + k4H,
      kH, kH, kH, k4H, 0, 0, 0, 0, 0);
  lstm_scan_kernel<<<dim3(4), dim3(512), 0, stream>>>(
      xg, whh_bf + (size_t)k4H * kH, h0 + kB * kH, c0 + kB * kH,
      cat_bf, 2 * kH);

  // ---- attention ----
  // keys = enc @ W_attn^T + b_attn
  gemm_nt_kernel<<<dim3(kH / 128, kBT / 128, 1), blk256, 0, stream>>>(
      enc_bf, wattn_bf, nullptr, keys_bf, b_attn, nullptr,
      kH, kH, kH, kH, 0, 0, 0, 0, 0);
  // scores[b] = out[b] @ keys[b]^T   (A rows live in cat buffer, lda=512)
  gemm_nt_kernel<<<dim3(kT / 128, kT / 128, kB), blk256, 0, stream>>>(
      cat_bf, keys_bf, scores, nullptr, nullptr, nullptr,
      kH, 2 * kH, kH, kT,
      (long long)kT * 2 * kH, (long long)kT * kH, (long long)kT * kT, 0, 0);
  softmax_kernel<<<dim3(kBT / 8), blk256, 0, stream>>>(scores, probs_bf);
  // ctx[b] = probs[b] @ enc[b]  == probs[b] @ (encT[b])^T, into right half of cat
  gemm_nt_kernel<<<dim3(kH / 128, kT / 128, kB), blk256, 0, stream>>>(
      probs_bf, encT_bf, nullptr, cat_bf + kH, nullptr, nullptr,
      kT, kT, kT, 2 * kH,
      (long long)kT * kT, (long long)kH * kT, 0, (long long)kT * 2 * kH, 0);

  // ---- head ----
  // outc = tanh(cat @ W_concat^T + b_concat)
  gemm_nt_kernel<<<dim3(kH / 128, kBT / 128, 1), blk256, 0, stream>>>(
      cat_bf, wcat_bf, nullptr, outc_bf, b_concat, nullptr,
      2 * kH, 2 * kH, 2 * kH, kH, 0, 0, 0, 0, 1);
  // logits = outc @ W_out^T  (f32 straight to d_out + bf16 copy for critic)
  gemm_nt_kernel<<<dim3(kO / 128, kBT / 128, 1), blk256, 0, stream>>>(
      outc_bf, wout_bf, logits_out, logits_bf, nullptr, nullptr,
      kH, kH, kH, kO, 0, 0, 0, 0, 0);
  // q = logits @ W_critic^T + b_critic
  gemm_nt_kernel<<<dim3(kO / 128, kBT / 128, 1), blk256, 0, stream>>>(
      logits_bf, wcrit_bf, qbuf, nullptr, b_critic, nullptr,
      kO, kO, kO, kO, 0, 0, 0, 0, 0);
  values_kernel<<<dim3(kBT / 8), blk256, 0, stream>>>(qbuf, logits_out, values_out);

  // ---- passthrough action output ----
  actions_to_f32_kernel<<<cdiv(kBT, 256), blk256, 0, stream>>>(
      actions, action_out, kBT);
}
